// SARANAttentionLayer_80118319940146
// MI455X (gfx1250) — matrix operand order repaired and checked by reading, not compiled, not run on hardware
//
#include <hip/hip_runtime.h>
#include <hip/hip_bf16.h>

// ---------------------------------------------------------------------------
// Causal full-embedding self-attention for gfx1250 (MI455X), bf16 WMMA path.
//   q = x @ Wq^T ; k = x @ Wk^T ; v = x @ Wv^T
//   out = x + softmax_causal(q k^T / sqrt(C)) v
// B=8, T=2048, C=640.
// ---------------------------------------------------------------------------

#define BATCH 8
#define T_SEQ 2048
#define C_DIM 640
#define SCALE 0.039528470752104744f  // 1/sqrt(640)

typedef __attribute__((ext_vector_type(16))) __bf16 bf16x16;
typedef __attribute__((ext_vector_type(8)))  __bf16 bf16x8;
typedef __attribute__((ext_vector_type(4)))  __bf16 bf16x4;
typedef __attribute__((ext_vector_type(8)))  float  f32x8;

__device__ __forceinline__ f32x8 wmma_bf16(bf16x16 a, bf16x16 b, f32x8 c) {
  // D = A(16x32 bf16) * B(32x16 bf16) + C(16x16 f32)
  return __builtin_amdgcn_wmma_f32_16x16x32_bf16(
      /*neg_a=*/false, a, /*neg_b=*/false, b,
      /*c_mod=*/(short)0, c, /*reuse_a=*/false, /*reuse_b=*/false);
}

// Load a 16-element bf16 fragment as two contiguous 16-byte chunks.
__device__ __forceinline__ bf16x16 ld2(const __bf16* p0, const __bf16* p1) {
  bf16x8 lo = *reinterpret_cast<const bf16x8*>(p0);
  bf16x8 hi = *reinterpret_cast<const bf16x8*>(p1);
  return __builtin_shufflevector(lo, hi, 0, 1, 2, 3, 4, 5, 6, 7,
                                         8, 9, 10, 11, 12, 13, 14, 15);
}

__device__ __forceinline__ f32x8 zero8() {
  f32x8 z = {0.f, 0.f, 0.f, 0.f, 0.f, 0.f, 0.f, 0.f};
  return z;
}

// ---------------------------------------------------------------------------
// fp32 -> bf16 conversion, 4-wide.
// ---------------------------------------------------------------------------
__global__ void cvt_f32_bf16_x4(const float* __restrict__ src,
                                __bf16* __restrict__ dst, int n4) {
  int i = blockIdx.x * blockDim.x + threadIdx.x;
  if (i >= n4) return;
  float4 v = reinterpret_cast<const float4*>(src)[i];
  bf16x4 o = {(__bf16)v.x, (__bf16)v.y, (__bf16)v.z, (__bf16)v.w};
  reinterpret_cast<bf16x4*>(dst)[i] = o;
}

// ---------------------------------------------------------------------------
// GEMM: Y[m,n] = sum_c A[m,c] * W[n,c]   (A: [B*T, C] bf16, W: [C,C] bf16,
// W in torch (out,in) layout -> W rows are columns of the WMMA B operand).
// Each wave computes a 16x64 tile (4 accumulators): the A fragment is reused
// across 4 WMMAs, cutting VMEM traffic from 4 to 2.5 b128-loads per WMMA.
// mode 0: Y stored row-major bf16 [B*T, C]  (used for Q and K)
// mode 1: Y stored transposed per batch     (used for V: Vt[b][n][t])
// ---------------------------------------------------------------------------
__global__ __launch_bounds__(128) void gemm_qkv(
    const __bf16* __restrict__ A, const __bf16* __restrict__ W,
    __bf16* __restrict__ Y, int transposed) {
  const int wave = threadIdx.x >> 5;
  const int lane = threadIdx.x & 31;
  const int ln = lane & 15;
  const int h = lane >> 4;

  const int NT = C_DIM / 64;  // 10 column-tiles of 64
  const int tile = blockIdx.x * 4 + wave;
  if (tile >= (BATCH * T_SEQ / 16) * NT) return;
  const int m0 = (tile / NT) * 16;
  const int n0 = (tile % NT) * 64;

  // A fragment base: row m0+ln, K chunks at +kbase and +kbase+8
  const __bf16* arow = A + (size_t)(m0 + ln) * C_DIM + h * 16;
  // B fragment base: column n0+sub*16+ln (a row of W), 16 K-contiguous elems
  const __bf16* wrow = W + (size_t)(n0 + ln) * C_DIM + h * 16;

  f32x8 acc[4];
#pragma unroll
  for (int s = 0; s < 4; ++s) acc[s] = zero8();

#pragma unroll
  for (int kc = 0; kc < C_DIM / 32; ++kc) {
    bf16x16 af = ld2(arow + kc * 32, arow + kc * 32 + 8);
#pragma unroll
    for (int s = 0; s < 4; ++s) {
      const __bf16* wp = wrow + (size_t)s * 16 * C_DIM + kc * 32;
      bf16x16 bf = ld2(wp, wp + 8);
      acc[s] = wmma_bf16(af, bf, acc[s]);
    }
  }

  if (!transposed) {
#pragma unroll
    for (int s = 0; s < 4; ++s)
#pragma unroll
      for (int r = 0; r < 8; ++r) {
        const int m = m0 + r + 8 * h;
        Y[(size_t)m * C_DIM + n0 + s * 16 + ln] = (__bf16)acc[s][r];
      }
  } else {
#pragma unroll
    for (int s = 0; s < 4; ++s)
#pragma unroll
      for (int r = 0; r < 8; ++r) {
        const int m = m0 + r + 8 * h;
        const int bb = m >> 11;          // / T_SEQ
        const int tt = m & (T_SEQ - 1);  // % T_SEQ
        Y[((size_t)bb * C_DIM + n0 + s * 16 + ln) * T_SEQ + tt] =
            (__bf16)acc[s][r];
      }
  }
}

// ---------------------------------------------------------------------------
// Flash attention. Block = 4 waves = one 16-query tile of one batch.
// Wave w owns channel slice [w*160, w*160+160).
// Per 64-key block (2 barriers per 40 WMMAs/wave):
//   - each wave: partial scores over its slice (4 subtiles x 5 WMMAs) -> LDS
//   - wave 0: sum partials, scale, causal mask, online softmax (shfl-xor row
//             reductions across 16-lane halves), write P bf16 + per-row alpha
//   - each wave: rescale its 16x160 f32 O accumulator, 2 P*V WMMAs per
//     channel tile (10 tiles)
// Epilogue: out = x + O / l.
// ---------------------------------------------------------------------------
__global__ __launch_bounds__(128) void attn_fwd(
    const __bf16* __restrict__ Qb,  // [B*T, C] bf16
    const __bf16* __restrict__ Kb,  // [B*T, C] bf16
    const __bf16* __restrict__ Vt,  // [B, C, T] bf16
    const float* __restrict__ x,    // [B, T, C] f32
    float* __restrict__ out) {      // [B, T, C] f32
  __shared__ float s_part[4][16][64];  // 16 KB partial scores
  __shared__ __bf16 s_p[16][64];       // 2 KB probabilities
  __shared__ float s_alpha[16];
  __shared__ float s_linv[16];

  const int wave = threadIdx.x >> 5;
  const int lane = threadIdx.x & 31;
  const int ln = lane & 15;
  const int h = lane >> 4;

  const int b = blockIdx.x >> 7;           // T/16 = 128 q-tiles per batch
  const int q0 = (blockIdx.x & 127) << 4;
  const int cslice = wave * 160;

  // Preload this wave's 5 Q fragments (16 queries x 32 channels each).
  bf16x16 qf[5];
  {
    const __bf16* qrow =
        Qb + (size_t)(b * T_SEQ + q0 + ln) * C_DIM + cslice + h * 16;
#pragma unroll
    for (int j = 0; j < 5; ++j)
      qf[j] = ld2(qrow + j * 32, qrow + j * 32 + 8);
  }

  f32x8 oacc[10];
#pragma unroll
  for (int t = 0; t < 10; ++t) oacc[t] = zero8();

  float m_run[8], l_run[8];
#pragma unroll
  for (int r = 0; r < 8; ++r) {
    m_run[r] = -INFINITY;
    l_run[r] = 0.f;
  }

  const int nkb = (q0 + 15) / 64 + 1;  // causal: need keys 0 .. q0+15
  for (int kb = 0; kb < nkb; ++kb) {
    const int k0 = kb * 64;

    // ---- partial scores over this wave's channel slice ----
#pragma unroll
    for (int sub = 0; sub < 4; ++sub) {
      f32x8 acc = zero8();
      const __bf16* kcol = Kb +
          (size_t)(b * T_SEQ + k0 + sub * 16 + ln) * C_DIM + cslice + h * 16;
#pragma unroll
      for (int j = 0; j < 5; ++j) {
        bf16x16 bf = ld2(kcol + j * 32, kcol + j * 32 + 8);
        acc = wmma_bf16(qf[j], bf, acc);
      }
#pragma unroll
      for (int r = 0; r < 8; ++r)
        s_part[wave][r + 8 * h][sub * 16 + ln] = acc[r];
    }
    __syncthreads();

    // ---- wave 0: reduce partials + online softmax ----
    if (wave == 0) {
      float pv[4][8], mc[8];
#pragma unroll
      for (int r = 0; r < 8; ++r) {
        const int row = r + 8 * h;
        const int q = q0 + row;
        float t = -INFINITY;
#pragma unroll
        for (int sub = 0; sub < 4; ++sub) {
          const int col = sub * 16 + ln;
          float v = s_part[0][row][col] + s_part[1][row][col] +
                    s_part[2][row][col] + s_part[3][row][col];
          v *= SCALE;
          if (k0 + col > q) v = -INFINITY;
          pv[sub][r] = v;
          t = fmaxf(t, v);
        }
#pragma unroll
        for (int msk = 1; msk < 16; msk <<= 1)
          t = fmaxf(t, __shfl_xor(t, msk, 32));
        mc[r] = t;  // block row-max, uniform across the 16-lane half
      }
#pragma unroll
      for (int r = 0; r < 8; ++r) {
        const int row = r + 8 * h;
        const float mnew = fmaxf(m_run[r], mc[r]);
        const float alpha = __expf(m_run[r] - mnew);
        float s = 0.f;
#pragma unroll
        for (int sub = 0; sub < 4; ++sub) {
          const float e = __expf(pv[sub][r] - mnew);
          s_p[row][sub * 16 + ln] = (__bf16)e;
          s += e;
        }
#pragma unroll
        for (int msk = 1; msk < 16; msk <<= 1) s += __shfl_xor(s, msk, 32);
        l_run[r] = l_run[r] * alpha + s;
        m_run[r] = mnew;
        if (ln == 0) s_alpha[row] = alpha;
      }
    }
    __syncthreads();

    // ---- all waves: rescale O, accumulate P * V over 10 channel tiles ----
    float a8[8];
#pragma unroll
    for (int r = 0; r < 8; ++r) a8[r] = s_alpha[r + 8 * h];

    const __bf16* pp0 = &s_p[ln][h * 16];        // keys k0 .. k0+31
    const __bf16* pp1 = &s_p[ln][32 + h * 16];   // keys k0+32 .. k0+63
    bf16x16 pa0 = ld2(pp0, pp0 + 8);
    bf16x16 pa1 = ld2(pp1, pp1 + 8);

    const __bf16* vcol =
        Vt + (size_t)(b * C_DIM + cslice + ln) * T_SEQ + k0 + h * 16;
#pragma unroll
    for (int t = 0; t < 10; ++t) {
#pragma unroll
      for (int r = 0; r < 8; ++r) oacc[t][r] *= a8[r];
      const __bf16* vp = vcol + (size_t)t * 16 * T_SEQ;
      bf16x16 vf0 = ld2(vp, vp + 8);
      bf16x16 vf1 = ld2(vp + 32, vp + 40);
      oacc[t] = wmma_bf16(pa0, vf0, oacc[t]);
      oacc[t] = wmma_bf16(pa1, vf1, oacc[t]);
    }
  }

  if (wave == 0 && ln == 0) {
#pragma unroll
    for (int r = 0; r < 8; ++r) s_linv[r + 8 * h] = 1.f / l_run[r];
  }
  __syncthreads();

  float li[8];
#pragma unroll
  for (int r = 0; r < 8; ++r) li[r] = s_linv[r + 8 * h];

#pragma unroll
  for (int t = 0; t < 10; ++t) {
#pragma unroll
    for (int r = 0; r < 8; ++r) {
      const int q = q0 + r + 8 * h;
      const int c = cslice + t * 16 + ln;
      const size_t idx = (size_t)(b * T_SEQ + q) * C_DIM + c;
      out[idx] = x[idx] + oacc[t][r] * li[r];
    }
  }
}

// ---------------------------------------------------------------------------
// Host launcher. Workspace layout (bytes, 256-aligned, ~83 MB total):
//   xb  [B*T*C] bf16, wqb/wkb/wvb [C*C] bf16,
//   Qb/Kb [B*T*C] bf16, Vt [B*C*T] bf16
// ---------------------------------------------------------------------------
extern "C" void kernel_launch(void* const* d_in, const int* in_sizes, int n_in,
                              void* d_out, int out_size, void* d_ws,
                              size_t ws_size, hipStream_t stream) {
  (void)in_sizes; (void)n_in; (void)out_size; (void)ws_size;

  const float* x  = (const float*)d_in[0];
  const float* wq = (const float*)d_in[1];
  const float* wk = (const float*)d_in[2];
  const float* wv = (const float*)d_in[3];
  float* out = (float*)d_out;

  const size_t NX = (size_t)BATCH * T_SEQ * C_DIM;  // 10,485,760
  const size_t NW = (size_t)C_DIM * C_DIM;          // 409,600

  char* ws = (char*)d_ws;
  size_t off = 0;
  auto carve = [&](size_t elems) {
    __bf16* p = (__bf16*)(ws + off);
    off = (off + elems * sizeof(__bf16) + 255) & ~(size_t)255;
    return p;
  };
  __bf16* xb  = carve(NX);
  __bf16* wqb = carve(NW);
  __bf16* wkb = carve(NW);
  __bf16* wvb = carve(NW);
  __bf16* Qb  = carve(NX);
  __bf16* Kb  = carve(NX);
  __bf16* Vt  = carve(NX);

  // Stage 1: fp32 -> bf16 conversions.
  {
    int n4 = (int)(NX / 4);
    cvt_f32_bf16_x4<<<(n4 + 255) / 256, 256, 0, stream>>>(x, xb, n4);
    int w4 = (int)(NW / 4);
    cvt_f32_bf16_x4<<<(w4 + 255) / 256, 256, 0, stream>>>(wq, wqb, w4);
    cvt_f32_bf16_x4<<<(w4 + 255) / 256, 256, 0, stream>>>(wk, wkb, w4);
    cvt_f32_bf16_x4<<<(w4 + 255) / 256, 256, 0, stream>>>(wv, wvb, w4);
  }

  // Stage 2: QKV projections (WMMA GEMM), one 16x64 tile per wave.
  {
    const int tiles = (BATCH * T_SEQ / 16) * (C_DIM / 64);  // 10240
    const int blocks = tiles / 4;                           // 2560
    gemm_qkv<<<blocks, 128, 0, stream>>>(xb, wqb, Qb, 0);
    gemm_qkv<<<blocks, 128, 0, stream>>>(xb, wkb, Kb, 0);
    gemm_qkv<<<blocks, 128, 0, stream>>>(xb, wvb, Vt, 1);  // V transposed
  }

  // Stage 3: causal flash attention + residual.
  attn_fwd<<<BATCH * (T_SEQ / 16), 128, 0, stream>>>(Qb, Kb, Vt, x, out);
}
